// GroupFNO2d_50311246905406
// MI455X (gfx1250) — matrix operand order repaired and checked by reading, not compiled, not run on hardware
//
#include <hip/hip_runtime.h>
#include <math.h>

typedef __attribute__((ext_vector_type(16))) _Float16 v16h;
typedef __attribute__((ext_vector_type(8)))  float    v8f;

#define G_    4
#define SC_   2
#define WID_  10
#define M1_   12
#define M2_   12
#define B_    8
#define H_    256
#define W_    256
#define C40_  (G_*WID_)        /* 40 */
#define KK_   (2*M2_)          /* 24 packed re/im freq columns */
#define ROWS_ (B_*C40_*H_)     /* 81920 */
#define NPIX_ (H_*W_)

static __device__ __forceinline__ v8f wmma_f16(v16h a, v16h b, v8f c) {
  // D = A(16x32 f16) * B(32x16 f16) + C(16x16 f32)
  return __builtin_amdgcn_wmma_f32_16x16x32_f16(false, a, false, b, (short)0, c, false, false);
}
// 16-bit A-matrix 16x32 layout: lane L (hi=L>>4): element e -> K
static __device__ __forceinline__ int a_kmap(int e, int hi) {
  return (e & 7) + ((e >> 3) << 4) + (hi << 3);
}
// CDNA5 async copy: per-lane 16B global -> LDS, tracked by ASYNCcnt.
static __device__ __forceinline__ void async_copy16(unsigned lds_byte_addr, const void* gaddr) {
  asm volatile("global_load_async_to_lds_b128 %0, %1, off"
               :: "v"(lds_byte_addr), "v"((unsigned long long)(size_t)gaddr)
               : "memory");
}
static __device__ __forceinline__ void async_wait0() {
  asm volatile("s_wait_asynccnt 0x0" ::: "memory");
}
static __device__ __forceinline__ void ds_wait0() {
  asm volatile("s_wait_dscnt 0x0" ::: "memory");
}

// ---------------------------------------------------------------------------
// DFT factor tables (built on device each launch; deterministic)
// FwF [256][32] f16 : forward W-DFT, col kk=2*ky+ri: re=cos, im=-sin (kk>=24 0)
// FhF [24][256][2] f32: forward H-DFT rows kx in {0..11,244..255}: e^{-i*}
// FhI [256][24][2] f32: inverse H-DFT with 1/(H*W) folded in: s*e^{+i*}
// WIm [32][256] f16 : inverse W-DFT, hermitian doubling, O(1) entries
// ---------------------------------------------------------------------------
__global__ void tables_kernel(_Float16* __restrict__ FwF, float* __restrict__ FhF,
                              float* __restrict__ FhI, _Float16* __restrict__ WIm) {
  int idx = blockIdx.x * blockDim.x + threadIdx.x;
  const float TP = 6.28318530717958647692f;
  if (idx < W_ * 32) {
    int w = idx >> 5, kk = idx & 31, ky = kk >> 1;
    float v = 0.f;
    if (ky < M2_) {
      float th = TP * (float)(w * ky) / (float)W_;
      v = (kk & 1) ? -sinf(th) : cosf(th);
    }
    FwF[idx] = (_Float16)v;
    return;
  }
  idx -= W_ * 32;
  if (idx < 24 * H_) {
    int j = idx / H_, h = idx % H_;
    int kx = (j < M1_) ? j : (H_ - 2 * M1_ + j);  // 244..255 for j=12..23
    float th = TP * (float)(kx * h) / (float)H_;
    FhF[idx * 2]     = cosf(th);
    FhF[idx * 2 + 1] = -sinf(th);
    return;
  }
  idx -= 24 * H_;
  if (idx < H_ * 24) {
    int h = idx / 24, j = idx % 24;
    int kx = (j < M1_) ? j : (H_ - 2 * M1_ + j);
    float th = TP * (float)(kx * h) / (float)H_;
    const float s = 1.f / (float)(H_ * W_);       // irfft2 normalization
    FhI[idx * 2]     = s * cosf(th);
    FhI[idx * 2 + 1] = s * sinf(th);
    return;
  }
  idx -= H_ * 24;
  if (idx < 32 * W_) {
    int kk = idx >> 8, w = idx & 255, ky = kk >> 1;
    float v = 0.f;
    if (kk == 0) v = 1.f;
    else if (kk == 1) v = 0.f;                    // Im(X0) never contributes
    else if (ky < M2_) {
      float th = TP * (float)(ky * w) / (float)W_;
      v = (kk & 1) ? (-2.f * sinf(th)) : (2.f * cosf(th));
    }
    WIm[idx] = (_Float16)v;
  }
}

// ---------------------------------------------------------------------------
// fc0: concat grid channels + grouped 1x1 conv (SC+2 -> WID per group), f16 out
// ---------------------------------------------------------------------------
__global__ void fc0_kernel(const float* __restrict__ x, const float* __restrict__ w,
                           const float* __restrict__ b, _Float16* __restrict__ h0) {
  int idx = blockIdx.x * blockDim.x + threadIdx.x;  // over B*H*W
  if (idx >= B_ * NPIX_) return;
  int wcol = idx % W_;
  int hrow = (idx / W_) % H_;
  int bb   = idx / NPIX_;
  float gx = (float)hrow * (1.f / (float)(H_ - 1));
  float gy = (float)wcol * (1.f / (float)(W_ - 1));
  int pix = hrow * W_ + wcol;
  for (int g = 0; g < G_; ++g) {
    float in0 = x[((size_t)(bb * (G_ * SC_) + g * SC_ + 0)) * NPIX_ + pix];
    float in1 = x[((size_t)(bb * (G_ * SC_) + g * SC_ + 1)) * NPIX_ + pix];
    #pragma unroll
    for (int o = 0; o < WID_; ++o) {
      const float* wp = w + (g * WID_ + o) * (SC_ + 2);
      float v = b[g * WID_ + o] + wp[0] * in0 + wp[1] * in1 + wp[2] * gx + wp[3] * gy;
      h0[((size_t)(bb * C40_ + g * WID_ + o)) * NPIX_ + pix] = (_Float16)v;
    }
  }
}

// ---------------------------------------------------------------------------
// Forward W-DFT as GEMM: Y1[row][kk] = sum_w h[row][w] * FwF[w][kk]
// 8 waves/WG, wave owns a 16-row M-tile; K=256 in 8 steps; N=24(pad 32).
// A panels staged via async global->LDS b128 (ASYNCcnt).
// ---------------------------------------------------------------------------
__global__ void dftw_kernel(const _Float16* __restrict__ src, const _Float16* __restrict__ FwF,
                            float* __restrict__ dst) {
  __shared__ _Float16 As[8][16][40];  // 80B rows: 16B aligned, conflict-benign
  int wv = threadIdx.x >> 5, lane = threadIdx.x & 31, l15 = lane & 15, hi = lane >> 4;
  size_t row0 = ((size_t)blockIdx.x * 8 + wv) * 16;
  v8f acc0 = {}, acc1 = {};
  for (int k0 = 0; k0 < W_; k0 += 32) {
    ds_wait0();  // WAR: previous step's LDS gathers complete before overwrite
    #pragma unroll
    for (int it = 0; it < 2; ++it) {            // 16 rows x 64B = 64 chunks of 16B
      int chunk = lane + it * 32;
      int r = chunk >> 2, sub = (chunk & 3) * 8;
      async_copy16((unsigned)(size_t)&As[wv][r][sub],
                   src + (row0 + r) * W_ + k0 + sub);
    }
    if (k0 + 32 < W_)
      __builtin_prefetch(src + row0 * W_ + k0 + 32, 0, 0);  // global_prefetch_b8
    async_wait0();
    v16h a, b0, b1;
    #pragma unroll
    for (int e = 0; e < 16; ++e)
      a[e] = As[wv][l15][a_kmap(e, hi)];
    const _Float16* Brow = FwF + (size_t)k0 * 32 + l15;  // lane = N column of B, elements over K
    #pragma unroll
    for (int e = 0; e < 16; ++e) { b0[e] = Brow[a_kmap(e, hi) * 32]; b1[e] = Brow[a_kmap(e, hi) * 32 + 16]; }
    acc0 = wmma_f16(a, b0, acc0);
    acc1 = wmma_f16(a, b1, acc1);
  }
  #pragma unroll
  for (int r = 0; r < 8; ++r) {
    int m = r + hi * 8;
    dst[(row0 + m) * KK_ + l15] = acc0[r];
    if (l15 < KK_ - 16) dst[(row0 + m) * KK_ + 16 + l15] = acc1[r];
  }
}

// ---------------------------------------------------------------------------
// Truncated H-DFT -> per-mode complex 10x10 channel mix -> inverse H-DFT
// (with 1/(H*W) folded into FhI). One WG per (b, g, ky); f32 VALU from LDS.
// ---------------------------------------------------------------------------
__global__ void spectral_kernel(const float* __restrict__ Y1, const float* __restrict__ FhF,
                                const float* __restrict__ FhI,
                                const float* __restrict__ sw1, const float* __restrict__ sw2,
                                int layer, float* __restrict__ Zf) {
  __shared__ float Yre[WID_][H_], Yim[WID_][H_];       // 20KB
  __shared__ float Mre[24][WID_], Mim[24][WID_];
  __shared__ float More[24][WID_], Moim[24][WID_];
  int bg = blockIdx.x / M2_, ky = blockIdx.x % M2_;
  int bb = bg / G_, gg = bg % G_;
  int t = threadIdx.x;  // 256 threads

  for (int i = 0; i < WID_; ++i) {
    size_t rbase = ((size_t)(bb * C40_ + gg * WID_ + i) * H_ + t) * KK_ + 2 * ky;
    Yre[i][t] = Y1[rbase];
    Yim[i][t] = Y1[rbase + 1];
  }
  __syncthreads();

  if (t < 24 * WID_) {  // forward H-DFT: 24 modes x 10 channels
    int j = t / WID_, i = t % WID_;
    float ar = 0.f, ai = 0.f;
    const float* F = FhF + (size_t)j * H_ * 2;
    for (int h = 0; h < H_; ++h) {
      float fr = F[2 * h], fi = F[2 * h + 1];
      float yr = Yre[i][h], yi = Yim[i][h];
      ar += fr * yr - fi * yi;
      ai += fr * yi + fi * yr;
    }
    Mre[j][i] = ar; Mim[j][i] = ai;
  }
  __syncthreads();

  if (t < 24 * WID_) {  // complex channel mixing with sp_w1 / sp_w2
    int j = t / WID_, o = t % WID_;
    int xm = (j < M1_) ? j : (j - M1_);
    const float* sw = (j < M1_) ? sw1 : sw2;
    float ar = 0.f, ai = 0.f;
    #pragma unroll
    for (int i = 0; i < WID_; ++i) {
      size_t widx = (((((size_t)layer * G_ + gg) * WID_ + i) * WID_ + o) * M1_ + xm) * M2_ + ky;
      float wr = sw[widx * 2], wi = sw[widx * 2 + 1];
      float mr = Mre[j][i], mi = Mim[j][i];
      ar += mr * wr - mi * wi;
      ai += mr * wi + mi * wr;
    }
    More[j][o] = ar; Moim[j][o] = ai;
  }
  __syncthreads();

  {  // inverse H-DFT (scaled): t = output row h
    int h = t;
    const float* F = FhI + (size_t)h * 24 * 2;
    for (int o = 0; o < WID_; ++o) {
      float ar = 0.f, ai = 0.f;
      #pragma unroll
      for (int j = 0; j < 24; ++j) {
        float fr = F[2 * j], fi = F[2 * j + 1];
        float mr = More[j][o], mi = Moim[j][o];
        ar += fr * mr - fi * mi;
        ai += fr * mi + fi * mr;
      }
      size_t rbase = ((size_t)(bb * C40_ + gg * WID_ + o) * H_ + h) * KK_ + 2 * ky;
      Zf[rbase] = ar;
      Zf[rbase + 1] = ai;
    }
  }
}

// ---------------------------------------------------------------------------
// Fused: inverse W-DFT (K=24 pad 32) + grouped 1x1 conv (K=10 pad 32)
//        + bias + GroupSwish. One WG per (b, g, row). Two chained WMMAs per
//        16-col tile share one f32 accumulator. Channel rows async-staged.
// ---------------------------------------------------------------------------
__global__ void invw_conv_kernel(const float* __restrict__ Zf, const _Float16* __restrict__ hin,
                                 const _Float16* __restrict__ WIm, const float* __restrict__ ww,
                                 const float* __restrict__ wb, const float* __restrict__ betas,
                                 int layer, int do_swish, _Float16* __restrict__ hout) {
  __shared__ float    Az[16][33];       // spectral A (10x24 zero-padded)
  __shared__ float    Wc[16][33];       // conv weights (10x10 zero-padded)
  __shared__ _Float16 Hs[WID_][264];    // conv B rows; 528B stride (16B aligned)
  int bb  = blockIdx.x / (G_ * H_);
  int rem = blockIdx.x % (G_ * H_);
  int gg = rem / H_, hh = rem % H_;
  int t = threadIdx.x, wv = t >> 5, lane = t & 31, l15 = lane & 15, hi = lane >> 4;

  // async-stage 10 channel rows x 512B = 320 chunks of 16B
  {
    const _Float16* gbase = hin + ((size_t)(bb * C40_ + gg * WID_) * H_ + hh) * W_;
    int r = t >> 5, sub = (t & 31) * 8;
    async_copy16((unsigned)(size_t)&Hs[r][sub], gbase + (size_t)r * NPIX_ + sub);
    if (t < 64) {
      int c = 256 + t;
      int r2 = c >> 5, sub2 = (c & 31) * 8;
      async_copy16((unsigned)(size_t)&Hs[r2][sub2], gbase + (size_t)r2 * NPIX_ + sub2);
    }
  }
  for (int idx = t; idx < 16 * 32; idx += 256) {
    int r = idx >> 5, c = idx & 31;
    float vz = 0.f, vw = 0.f;
    if (r < WID_) {
      if (c < KK_)  vz = Zf[((size_t)(bb * C40_ + gg * WID_ + r) * H_ + hh) * KK_ + c];
      if (c < WID_) vw = ww[(((size_t)layer * G_ + gg) * WID_ + r) * WID_ + c];
    }
    Az[r][c] = vz; Wc[r][c] = vw;
  }
  async_wait0();
  __syncthreads();

  float bet = betas[layer * G_ + gg];
  float sp  = log1pf(expf(bet));  // softplus(beta)

  v16h az, aw;
  #pragma unroll
  for (int e = 0; e < 16; ++e) {
    int k = a_kmap(e, hi);
    az[e] = (_Float16)Az[l15][k];
    aw[e] = (_Float16)Wc[l15][k];
  }
  #pragma unroll
  for (int nt = wv * 2; nt < wv * 2 + 2; ++nt) {  // 16 N-tiles over 8 waves
    int w0 = nt * 16;
    v16h bs, bh;
    #pragma unroll
    for (int e = 0; e < 16; ++e) {
      int k = a_kmap(e, hi);
      bs[e] = WIm[(size_t)k * W_ + w0 + l15];                    // L2-hot table
      bh[e] = (k < WID_) ? Hs[k][w0 + l15] : (_Float16)0.f;   // K-pad in gather
    }
    v8f acc = {};
    acc = wmma_f16(az, bs, acc);   // spectral branch
    acc = wmma_f16(aw, bh, acc);   // conv branch accumulates on top
    #pragma unroll
    for (int r = 0; r < 8; ++r) {
      int o = r + hi * 8;
      if (o < WID_) {
        float v = acc[r] + wb[layer * C40_ + gg * WID_ + o];
        if (do_swish) v = v * (1.f / (1.f + expf(-v * sp))) * (1.f / 1.1f);
        hout[((size_t)(bb * C40_ + gg * WID_ + o) * H_ + hh) * W_ + w0 + l15] = (_Float16)v;
      }
    }
  }
}

// ---------------------------------------------------------------------------
// Head: fc1 (10->128) + GroupSwish + fc2 (128->2), fused per (b, g, row).
// fc1: M=128 K=10(pad32) N=256; fc2: M=2(pad16) K=128, in 32-col chunks.
// ---------------------------------------------------------------------------
__global__ void head_kernel(const _Float16* __restrict__ hin, const float* __restrict__ fc1w,
                            const float* __restrict__ fc1b, const float* __restrict__ fc2w,
                            const float* __restrict__ fc2b, const float* __restrict__ betas,
                            float* __restrict__ out) {
  __shared__ _Float16 Hs[WID_][264];  // fc1 B rows (K-pad via gather predicate)
  __shared__ float    Ys[128][41];    // fc1 activation chunk (32 cols + pad)
  __shared__ float    A2[16][129];    // fc2 weights padded to M=16
  int bb  = blockIdx.x / (G_ * H_);
  int rem = blockIdx.x % (G_ * H_);
  int gg = rem / H_, hh = rem % H_;
  int t = threadIdx.x, wv = t >> 5, lane = t & 31, l15 = lane & 15, hi = lane >> 4;

  {
    const _Float16* gbase = hin + ((size_t)(bb * C40_ + gg * WID_) * H_ + hh) * W_;
    int r = t >> 5, sub = (t & 31) * 8;
    async_copy16((unsigned)(size_t)&Hs[r][sub], gbase + (size_t)r * NPIX_ + sub);
    if (t < 64) {
      int c = 256 + t;
      int r2 = c >> 5, sub2 = (c & 31) * 8;
      async_copy16((unsigned)(size_t)&Hs[r2][sub2], gbase + (size_t)r2 * NPIX_ + sub2);
    }
  }
  for (int idx = t; idx < 16 * 128; idx += 256) {
    int r = idx >> 7, k = idx & 127;
    A2[r][k] = (r < SC_) ? fc2w[((size_t)gg * SC_ + r) * 128 + k] : 0.f;
  }
  async_wait0();
  __syncthreads();

  float bet = betas[3 * G_ + gg];
  float sp  = log1pf(expf(bet));

  // fc1 A fragment: wave wv owns M-tile wv (rows wv*16..wv*16+15), single K step
  v16h a1;
  #pragma unroll
  for (int e = 0; e < 16; ++e) {
    int k = a_kmap(e, hi);
    a1[e] = (k < WID_) ? (_Float16)fc1w[((size_t)gg * 128 + wv * 16 + l15) * WID_ + k]
                       : (_Float16)0.f;
  }

  for (int chunk = 0; chunk < 8; ++chunk) {
    int wbase = chunk * 32;
    #pragma unroll
    for (int nt = 0; nt < 2; ++nt) {
      int w0 = wbase + nt * 16;
      v16h bfr;
      #pragma unroll
      for (int e = 0; e < 16; ++e)
        bfr[e] = (a_kmap(e, hi) < WID_) ? Hs[a_kmap(e, hi)][w0 + l15] : (_Float16)0.f;
      v8f acc = {};
      acc = wmma_f16(a1, bfr, acc);
      #pragma unroll
      for (int r = 0; r < 8; ++r) {
        int o = wv * 16 + r + hi * 8;
        float v = acc[r] + fc1b[gg * 128 + o];
        v = v * (1.f / (1.f + expf(-v * sp))) * (1.f / 1.1f);
        Ys[o][nt * 16 + l15] = v;
      }
    }
    __syncthreads();
    if (wv < 2) {  // fc2 over this 32-col chunk: 2 N-tiles, K=128 in 4 steps
      v8f acc = {};
      int w0l = wv * 16;
      #pragma unroll
      for (int k0 = 0; k0 < 128; k0 += 32) {
        v16h a2, b2;
        #pragma unroll
        for (int e = 0; e < 16; ++e) {
          a2[e] = (_Float16)A2[l15][k0 + a_kmap(e, hi)];
          b2[e] = (_Float16)Ys[k0 + a_kmap(e, hi)][w0l + l15];
        }
        acc = wmma_f16(a2, b2, acc);
      }
      #pragma unroll
      for (int r = 0; r < 8; ++r) {
        int s = r + hi * 8;
        if (s < SC_) {
          int w = wbase + w0l + l15;
          out[((size_t)(bb * G_ * SC_ + gg * SC_ + s) * H_ + hh) * W_ + w] =
              acc[r] + fc2b[gg * SC_ + s];
        }
      }
    }
    __syncthreads();
  }
}

// ---------------------------------------------------------------------------
extern "C" void kernel_launch(void* const* d_in, const int* in_sizes, int n_in,
                              void* d_out, int out_size, void* d_ws, size_t ws_size,
                              hipStream_t stream) {
  (void)in_sizes; (void)n_in; (void)out_size; (void)ws_size;
  const float* x     = (const float*)d_in[0];
  const float* fc0w  = (const float*)d_in[1];
  const float* fc0b  = (const float*)d_in[2];
  const float* ww    = (const float*)d_in[3];
  const float* wb    = (const float*)d_in[4];
  const float* sw1   = (const float*)d_in[5];
  const float* sw2   = (const float*)d_in[6];
  const float* betas = (const float*)d_in[7];
  const float* fc1w  = (const float*)d_in[8];
  const float* fc1b  = (const float*)d_in[9];
  const float* fc2w  = (const float*)d_in[10];
  const float* fc2b  = (const float*)d_in[11];

  const size_t HSZ = (size_t)B_ * C40_ * NPIX_;  // 20,971,520 f16 elements
  const size_t FSZ = (size_t)ROWS_ * KK_;        //  1,966,080 f32 elements
  _Float16* hA  = (_Float16*)d_ws;
  _Float16* hB  = hA + HSZ;
  float*    Y1  = (float*)(hB + HSZ);
  float*    Zf  = Y1 + FSZ;
  _Float16* FwF = (_Float16*)(Zf + FSZ);
  float*    FhF = (float*)(FwF + (size_t)W_ * 32);
  float*    FhI = FhF + (size_t)24 * H_ * 2;
  _Float16* WIm = (_Float16*)(FhI + (size_t)H_ * 24 * 2);

  tables_kernel<<<112, 256, 0, stream>>>(FwF, FhF, FhI, WIm);
  fc0_kernel<<<(B_ * NPIX_) / 256, 256, 0, stream>>>(x, fc0w, fc0b, hA);

  _Float16* cur = hA;
  _Float16* nxt = hB;
  for (int l = 0; l < 4; ++l) {
    dftw_kernel<<<ROWS_ / 128, 256, 0, stream>>>(cur, FwF, Y1);
    spectral_kernel<<<B_ * G_ * M2_, 256, 0, stream>>>(Y1, FhF, FhI, sw1, sw2, l, Zf);
    invw_conv_kernel<<<B_ * G_ * H_, 256, 0, stream>>>(Zf, cur, WIm, ww, wb, betas,
                                                       l, (l < 3) ? 1 : 0, nxt);
    _Float16* tmp = cur; cur = nxt; nxt = tmp;
  }
  head_kernel<<<B_ * G_ * H_, 256, 0, stream>>>(cur, fc1w, fc1b, fc2w, fc2b, betas,
                                                (float*)d_out);
}
